// SelfAttention_15144054686429
// MI455X (gfx1250) — matrix operand order repaired and checked by reading, not compiled, not run on hardware
//
#include <hip/hip_runtime.h>

typedef __attribute__((ext_vector_type(16))) _Float16 v16h;
typedef __attribute__((ext_vector_type(8)))  _Float16 v8h;
typedef __attribute__((ext_vector_type(8)))  float    v8f;

constexpr int D_MODEL = 1024;
constexpr int NSEQ    = 2048;
constexpr int BATCH   = 2;
constexpr int NHEAD   = 16;
constexpr int DK      = 64;
constexpr int ROWS    = BATCH * NSEQ;   // 4096

// ---------------------------------------------------------------------------
// Operand loaders with contiguous k order (same for A and B; not the ISA table order)
//
// A (16x32 f16), row-major source [16 rows, ld]:
//   lane L: M = L%16; halfs i -> K = (L>=16)*16 + i
//   => one contiguous 32-byte load per lane.
// B (32x16 f16), "N-major" source [16 rows = N, ld, K contiguous]:
//   lane L: N = L%16; halfs i -> K = (L>=16)*16 + i
//   => one contiguous 32-byte load per lane.
// ---------------------------------------------------------------------------
__device__ inline v16h load_a16(const _Float16* base, int ld) {
    int lane = threadIdx.x & 31;
    int r = lane & 15;
    int g = (lane >> 4) & 1;
    return *(const v16h*)(base + (size_t)r * ld + g * 16);
}

__device__ inline v16h load_b16(const _Float16* base, int ld) {
    int lane = threadIdx.x & 31;
    int n = lane & 15;
    int g = (lane >> 4) & 1;
    return *(const v16h*)(base + (size_t)n * ld + g * 16);
}

// ---------------------------------------------------------------------------
// 1) fp32 -> fp16 streaming conversion
// ---------------------------------------------------------------------------
__global__ void __launch_bounds__(256) cvt_f32_f16(const float* __restrict__ in,
                                                   _Float16* __restrict__ out, int n) {
    int i = blockIdx.x * 256 + threadIdx.x;
    if (i < n) out[i] = (_Float16)in[i];
}

// ---------------------------------------------------------------------------
// 2) Fused QKV projection with 64x64 register tiling (16 WMMA chains/wave).
//    C[64x64] = Xh(64xK) * W^T(Kx64) + b, K = 1024.
//    blockIdx = (M-tile, N-tile, which ∈ {Q,K,V}); one wave per tile.
//    Q pre-scaled by 1/sqrt(dk)=0.125 (exact in f16).
//    Q,K stored [b,h,seq,dk]; V stored transposed [b,h,dk,seq].
// ---------------------------------------------------------------------------
__global__ void __launch_bounds__(32) qkv_proj(
    const _Float16* __restrict__ Xh,
    const _Float16* __restrict__ Wqh, const _Float16* __restrict__ Wkh,
    const _Float16* __restrict__ Wvh,
    const float* __restrict__ bq, const float* __restrict__ bk,
    const float* __restrict__ bv,
    _Float16* __restrict__ Qh, _Float16* __restrict__ Kh,
    _Float16* __restrict__ VhT) {
    int lane = threadIdx.x;
    int mt = blockIdx.x, nt = blockIdx.y, z = blockIdx.z;
    const _Float16* W    = (z == 0) ? Wqh : (z == 1) ? Wkh : Wvh;
    const float*    bias = (z == 0) ? bq  : (z == 1) ? bk  : bv;
    int m0 = mt * 64, n0 = nt * 64;

    v8f acc[4][4] = {};   // [m-subtile][n-subtile], 16 independent chains
#pragma unroll 2
    for (int kk = 0; kk < D_MODEL / 32; ++kk) {
        v16h a[4], bb[4];
#pragma unroll
        for (int mm = 0; mm < 4; ++mm)
            a[mm] = load_a16(Xh + (size_t)(m0 + mm * 16) * D_MODEL + kk * 32, D_MODEL);
#pragma unroll
        for (int nn = 0; nn < 4; ++nn)
            bb[nn] = load_b16(W + (size_t)(n0 + nn * 16) * D_MODEL + kk * 32, D_MODEL);
#pragma unroll
        for (int mm = 0; mm < 4; ++mm)
#pragma unroll
            for (int nn = 0; nn < 4; ++nn)
                acc[mm][nn] = __builtin_amdgcn_wmma_f32_16x16x32_f16(
                    false, a[mm], false, bb[nn], (short)0, acc[mm][nn], false, false);
    }

    int hi = lane >> 4, nl = lane & 15;
    int batch = m0 >> 11;   // m0 / 2048 (64-row tiles never straddle batches)
    int seq0 = m0 & 2047;

#pragma unroll
    for (int nn = 0; nn < 4; ++nn) {
        int nfeat = n0 + nn * 16 + nl;
        float bval = bias[nfeat];
        int hh = nfeat >> 6;   // head
        int dk = nfeat & 63;
        size_t bhh = (size_t)batch * NHEAD + hh;
#pragma unroll
        for (int mm = 0; mm < 4; ++mm) {
#pragma unroll
            for (int j = 0; j < 8; ++j) {
                int seq = seq0 + mm * 16 + j + 8 * hi;  // C layout: VGPR j, hi-half -> +8
                float v = acc[mm][nn][j] + bval;
                if (z == 0)
                    Qh[(bhh * NSEQ + seq) * DK + dk] = (_Float16)(v * 0.125f);
                else if (z == 1)
                    Kh[(bhh * NSEQ + seq) * DK + dk] = (_Float16)v;
                else
                    VhT[(bhh * DK + dk) * (size_t)NSEQ + seq] = (_Float16)v;
            }
        }
    }
}

// ---------------------------------------------------------------------------
// 3) Flash attention: one wave per (b, h, 32-query tile).
//    Streams keys in 32-wide chunks; online softmax in fp32; P staged through
//    LDS for the C-layout -> A-layout conversion; O accumulated 32x64 in fp32.
//    Each K/V B-operand load now feeds 2 WMMAs (two query subtiles).
// ---------------------------------------------------------------------------
constexpr int QT = 2;   // query subtiles (16 rows each) per wave

__global__ void __launch_bounds__(32) flash_attn(
    const _Float16* __restrict__ Qh, const _Float16* __restrict__ Kh,
    const _Float16* __restrict__ VhT, float* __restrict__ out) {
    __shared__ _Float16 Plds[QT * 16 * 32];  // P tiles, row-major [16 rows][32 keys]

    int lane = threadIdx.x;
    int qblk = blockIdx.x;    // 0..63  (32 queries each)
    int h = blockIdx.y;       // 0..15
    int b = blockIdx.z;       // 0..1
    size_t bh = (size_t)b * NHEAD + h;
    const _Float16* Qbase = Qh  + bh * (size_t)NSEQ * DK;
    const _Float16* Kbase = Kh  + bh * (size_t)NSEQ * DK;
    const _Float16* Vbase = VhT + bh * (size_t)DK * NSEQ;
    int q0 = qblk * (QT * 16);

    // Q tiles (2 x 16x64) as A operands, reused across all key chunks
    v16h aq[QT][2];
#pragma unroll
    for (int qt = 0; qt < QT; ++qt) {
        aq[qt][0] = load_a16(Qbase + (size_t)(q0 + qt * 16) * DK, DK);
        aq[qt][1] = load_a16(Qbase + (size_t)(q0 + qt * 16) * DK + 32, DK);
    }

    float m_r[QT][8], l_r[QT][8];
    v8f acc[QT][4] = {};   // O tiles: 2 x (16x64)
#pragma unroll
    for (int qt = 0; qt < QT; ++qt)
#pragma unroll
        for (int j = 0; j < 8; ++j) { m_r[qt][j] = -1e30f; l_r[qt][j] = 0.0f; }

    int hi = lane >> 4, nl = lane & 15;

    for (int kc = 0; kc < NSEQ / 32; ++kc) {
        int key0 = kc * 32;
        // K operands for 32 keys: [key subtile][dk half]
        v16h bk[2][2];
#pragma unroll
        for (int st = 0; st < 2; ++st) {
            bk[st][0] = load_b16(Kbase + (size_t)(key0 + st * 16) * DK, DK);
            bk[st][1] = load_b16(Kbase + (size_t)(key0 + st * 16) * DK + 32, DK);
        }
        // S = Q * K^T : 8 WMMAs (2 q-subtiles x 2 key-subtiles x 2 dk-halves)
        v8f s[QT][2] = {};
#pragma unroll
        for (int qt = 0; qt < QT; ++qt)
#pragma unroll
            for (int st = 0; st < 2; ++st) {
                s[qt][st] = __builtin_amdgcn_wmma_f32_16x16x32_f16(
                    false, aq[qt][0], false, bk[st][0], (short)0, s[qt][st], false, false);
                s[qt][st] = __builtin_amdgcn_wmma_f32_16x16x32_f16(
                    false, aq[qt][1], false, bk[st][1], (short)0, s[qt][st], false, false);
            }

        // Online softmax. C layout: VGPR j holds row (j + 8*hi), 16 cols across
        // the 16-lane half -> row reductions via shfl_xor masks 1,2,4,8.
#pragma unroll
        for (int qt = 0; qt < QT; ++qt) {
#pragma unroll
            for (int j = 0; j < 8; ++j) {
                float v0 = s[qt][0][j], v1 = s[qt][1][j];
                float mx = fmaxf(v0, v1);
#pragma unroll
                for (int msk = 1; msk < 16; msk <<= 1)
                    mx = fmaxf(mx, __shfl_xor(mx, msk, 32));
                float mnew = fmaxf(m_r[qt][j], mx);
                float alpha = __expf(m_r[qt][j] - mnew);
                float p0 = __expf(v0 - mnew);
                float p1 = __expf(v1 - mnew);
                float rs = p0 + p1;
#pragma unroll
                for (int msk = 1; msk < 16; msk <<= 1)
                    rs += __shfl_xor(rs, msk, 32);
                l_r[qt][j] = l_r[qt][j] * alpha + rs;
                m_r[qt][j] = mnew;
#pragma unroll
                for (int n = 0; n < 4; ++n) acc[qt][n][j] *= alpha;
                int row = j + 8 * hi;
                Plds[qt * 512 + row * 32 + nl]      = (_Float16)p0;
                Plds[qt * 512 + row * 32 + 16 + nl] = (_Float16)p1;
            }
        }
        __syncthreads();
        v16h ap[QT];
#pragma unroll
        for (int qt = 0; qt < QT; ++qt)
            ap[qt] = load_a16(&Plds[qt * 512], 32);   // P as 16x32 A operand
        __syncthreads();

        // O += P * V : four dk-chunks; each V operand feeds both q-subtiles
#pragma unroll
        for (int n = 0; n < 4; ++n) {
            v16h bv = load_b16(Vbase + (size_t)(n * 16) * NSEQ + key0, NSEQ);
#pragma unroll
            for (int qt = 0; qt < QT; ++qt)
                acc[qt][n] = __builtin_amdgcn_wmma_f32_16x16x32_f16(
                    false, ap[qt], false, bv, (short)0, acc[qt][n], false, false);
        }
    }

    // Epilogue: O /= l, scatter to out[b, seq, h*64 + dk] (fp32)
#pragma unroll
    for (int qt = 0; qt < QT; ++qt) {
#pragma unroll
        for (int j = 0; j < 8; ++j) {
            int row = q0 + qt * 16 + j + 8 * hi;
            float inv = 1.0f / l_r[qt][j];
            size_t base = ((size_t)b * NSEQ + row) * D_MODEL + h * DK;
#pragma unroll
            for (int n = 0; n < 4; ++n)
                out[base + n * 16 + nl] = acc[qt][n][j] * inv;
        }
    }
}

// ---------------------------------------------------------------------------
// Launch
// ---------------------------------------------------------------------------
extern "C" void kernel_launch(void* const* d_in, const int* in_sizes, int n_in,
                              void* d_out, int out_size, void* d_ws, size_t ws_size,
                              hipStream_t stream) {
    const float* x  = (const float*)d_in[0];
    const float* Wq = (const float*)d_in[1];
    const float* bq = (const float*)d_in[2];
    const float* Wk = (const float*)d_in[3];
    const float* bk = (const float*)d_in[4];
    const float* Wv = (const float*)d_in[5];
    const float* bv = (const float*)d_in[6];
    float* out = (float*)d_out;

    // Workspace layout (bytes)
    char* ws = (char*)d_ws;
    const size_t xh_bytes  = (size_t)ROWS * D_MODEL * 2;            // 8 MiB
    const size_t w_bytes   = (size_t)D_MODEL * D_MODEL * 2;         // 2 MiB each
    const size_t qkv_bytes = (size_t)BATCH * NHEAD * NSEQ * DK * 2; // 8 MiB each
    _Float16* Xh  = (_Float16*)(ws);
    _Float16* Wqh = (_Float16*)(ws + xh_bytes);
    _Float16* Wkh = (_Float16*)(ws + xh_bytes + w_bytes);
    _Float16* Wvh = (_Float16*)(ws + xh_bytes + 2 * w_bytes);
    _Float16* Qh  = (_Float16*)(ws + xh_bytes + 3 * w_bytes);
    _Float16* Kh  = (_Float16*)(ws + xh_bytes + 3 * w_bytes + qkv_bytes);
    _Float16* VhT = (_Float16*)(ws + xh_bytes + 3 * w_bytes + 2 * qkv_bytes);

    // 1) f32 -> f16 conversions
    {
        int nX = ROWS * D_MODEL;
        cvt_f32_f16<<<(nX + 255) / 256, 256, 0, stream>>>(x, Xh, nX);
        int nW = D_MODEL * D_MODEL;
        cvt_f32_f16<<<(nW + 255) / 256, 256, 0, stream>>>(Wq, Wqh, nW);
        cvt_f32_f16<<<(nW + 255) / 256, 256, 0, stream>>>(Wk, Wkh, nW);
        cvt_f32_f16<<<(nW + 255) / 256, 256, 0, stream>>>(Wv, Wvh, nW);
    }

    // 2) QKV projection: grid (M-tiles, N-tiles, {q,k,v}), one wave per 64x64 tile
    {
        dim3 grid(ROWS / 64, D_MODEL / 64, 3);
        qkv_proj<<<grid, 32, 0, stream>>>(Xh, Wqh, Wkh, Wvh, bq, bk, bv,
                                          Qh, Kh, VhT);
    }

    // 3) Flash attention: one wave per (32-query tile, head, batch)
    {
        dim3 grid(NSEQ / (QT * 16), NHEAD, BATCH);
        flash_attn<<<grid, 32, 0, stream>>>(Qh, Kh, VhT, out);
    }
}
